// SsLayer_35880156791446
// MI455X (gfx1250) — matrix operand order repaired and checked by reading, not compiled, not run on hardware
//
#include <hip/hip_runtime.h>
#include <hip/hip_bf16.h>

typedef __attribute__((ext_vector_type(16))) __bf16 v16bf;
typedef __attribute__((ext_vector_type(8)))  __bf16 v8bf;
typedef __attribute__((ext_vector_type(8)))  float  v8f;
typedef unsigned int u32x4 __attribute__((ext_vector_type(4)));
typedef int          i32x4 __attribute__((ext_vector_type(4)));
typedef int          i32x8 __attribute__((ext_vector_type(8)));

#define BSZ   64
#define TLEN  1024
#define INDIM 256
#define NST   512
#define ODIM  256
#define KTOT  (INDIM + NST)   // 768
#define LDSS  520             // state row stride (halves): 512 + 8 -> conflict-free banks
#define XSTR  264             // x tile row stride (halves): 256 + 8 (TDM pad: 4 dwords/row)

#if __has_builtin(__builtin_amdgcn_tensor_load_to_lds) && __has_builtin(__builtin_amdgcn_s_wait_tensorcnt)
#define USE_TDM 1
#else
#define USE_TDM 0
#endif

// ---- one-time packs: bf16 weights (N-major) and bf16 copy of x ----
__global__ void pack_w_kernel(const float* __restrict__ A,
                              const float* __restrict__ Bm,
                              __bf16* __restrict__ Wt) {
  int idx = blockIdx.x * blockDim.x + threadIdx.x;   // over NST*KTOT
  if (idx >= NST * KTOT) return;
  int n = idx / KTOT;
  int k = idx - n * KTOT;
  float v = (k < INDIM) ? Bm[(size_t)k * NST + n] : A[(size_t)(k - INDIM) * NST + n];
  Wt[idx] = (__bf16)v;
}

__global__ void pack_cd_kernel(const float* __restrict__ C,
                               const float* __restrict__ D,
                               __bf16* __restrict__ CDt) {
  int idx = blockIdx.x * blockDim.x + threadIdx.x;   // over NST*NST
  if (idx >= NST * NST) return;
  int n = idx / NST;
  int k = idx - n * NST;
  float v = (n < ODIM) ? C[(size_t)k * ODIM + n] : D[(size_t)k * ODIM + (n - ODIM)];
  CDt[idx] = (__bf16)v;
}

__global__ void pack_x_kernel(const float* __restrict__ x, __bf16* __restrict__ Xbf) {
  size_t idx = (size_t)blockIdx.x * blockDim.x + threadIdx.x;  // over B*T*IN
  if (idx >= (size_t)BSZ * TLEN * INDIM) return;
  Xbf[idx] = (__bf16)x[idx];
}

__device__ __forceinline__ v16bf cat8(v8bf lo, v8bf hi) {
  return __builtin_shufflevector(lo, hi, 0,1,2,3,4,5,6,7,8,9,10,11,12,13,14,15);
}

#if USE_TDM
// TDM: DMA the 64x256 bf16 x_t tile into LDS, padding each 512B row by 16B so the
// LDS layout has row stride XSTR (bank-conflict-free WMMA fragment reads).
// D# packing per CDNA5 ISA.
__device__ __forceinline__ void tdm_load_xt(const __bf16* Xbf, int t, __bf16* Xs_ptr) {
  unsigned long long ga = (unsigned long long)(size_t)Xbf +
                          (unsigned long long)t * (INDIM * 2);  // tile start (bytes)
  unsigned lds = (unsigned)(size_t)Xs_ptr;                      // LDS byte offset
  u32x4 g0 = { 1u,                                              // count=1, user desc
               lds,                                             // lds_addr
               (unsigned)ga,                                    // global_addr[31:0]
               ((unsigned)((ga >> 32) & 0x01FFFFFFull)) | 0x80000000u }; // [56:32] | type=2
  // g1: data_size=2B (code 1), pad_enable, pad_interval=128 dwords (code 6),
  //     pad_amount=4 dwords (code 3)  -> d0 = 0x07910000
  //     tensor_dim0 = 0x40000 (=T*IN, elems), tensor_dim1 = 64,
  //     tile_dim0 = 256, tile_dim1 = 64, tensor_dim0_stride = 0x40000
  i32x8 g1 = { 0x07910000, 0x00000000, 0x00400004, 0x01000000,
               64, 0x40000, 0, 0 };
  i32x4 z4 = { 0, 0, 0, 0 };
#if defined(__clang_major__) && (__clang_major__ >= 23)
  i32x8 z8 = { 0, 0, 0, 0, 0, 0, 0, 0 };
  __builtin_amdgcn_tensor_load_to_lds(g0, g1, z4, z4, z8, 0);
#else
  __builtin_amdgcn_tensor_load_to_lds(g0, g1, z4, z4, 0);
#endif
}
#endif

// ---- persistent single-WGP recurrence kernel ----
__global__ __launch_bounds__(1024, 1)
void ssm_kernel(const __bf16* __restrict__ Xbf,
                const float* __restrict__ bvec,
                const float* __restrict__ cvec,
                const float* __restrict__ dvec,
                const __bf16* __restrict__ Wt,
                const __bf16* __restrict__ CDt,
                float* __restrict__ out) {
  __shared__ __bf16 Xs[BSZ * XSTR];      // x_t tile (bf16), TDM-staged, padded rows
  __shared__ __bf16 Ss[BSZ * LDSS];      // recurrent state (bf16)

  const int tid  = threadIdx.x;
  const int wave = tid >> 5;
  const int lane = tid & 31;
  const int hsel = lane >> 4;
  const int l16  = lane & 15;
  const int kb   = hsel * 8;             // A-frag K base / D-frag row base (ISA layout)

  // 2x2 tile block per wave: rows rp*32..+31, cols cp*32..+31
  const int rp = wave >> 4;              // 0..1  (row-block pair)
  const int cp = wave & 15;              // 0..15 (col-block pair)
  const int c0 = cp * 32;

  for (int i = tid; i < BSZ * LDSS; i += 1024) Ss[i] = (__bf16)0.0f;

  float biasb[2], biascd[2];
  #pragma unroll
  for (int cj = 0; cj < 2; ++cj) {
    const int col = c0 + cj * 16 + l16;
    biasb[cj]  = bvec[col];
    biascd[cj] = (c0 < ODIM) ? cvec[col] : dvec[col - ODIM];
  }
  float* stateout = out + (size_t)2 * BSZ * TLEN * ODIM;

  // stage x_0
#if USE_TDM
  if (wave == 0) tdm_load_xt(Xbf, 0, Xs);
#else
  {
    const int row = tid >> 4, k0c = (tid & 15) * 16;
    const __bf16* src = Xbf + (size_t)row * (TLEN * INDIM) + k0c;
    *(v8bf*)&Xs[row * XSTR + k0c]     = *(const v8bf*)src;
    *(v8bf*)&Xs[row * XSTR + k0c + 8] = *(const v8bf*)(src + 8);
  }
#endif
  __syncthreads();

  for (int t = 0; t < TLEN; ++t) {
#if USE_TDM
    if (wave == 0) __builtin_amdgcn_s_wait_tensorcnt(0);  // x_t DMA complete
#endif
    __syncthreads();                     // publish x_t to all waves

    // ---- GEMM1: P = [x_t ; S] @ [Bm ; A]  (64x768 @ 768x512) ----
    v8f acc[2][2] = {};
    #pragma unroll
    for (int kc = 0; kc < KTOT / 32; ++kc) {
      const int k0 = kc * 32;
      v16bf af[2];
      #pragma unroll
      for (int ri = 0; ri < 2; ++ri) {   // two A-frags, each reused by two WMMAs
        const int m = rp * 32 + ri * 16 + l16;
        const __bf16* sp = (k0 < INDIM) ? &Xs[m * XSTR + k0 + kb]
                                        : &Ss[m * LDSS + (k0 - INDIM) + kb];
        af[ri] = cat8(*(const v8bf*)sp, *(const v8bf*)(sp + 16));
      }
      #pragma unroll
      for (int cj = 0; cj < 2; ++cj) {
        const __bf16* bp = Wt + (size_t)(c0 + cj * 16 + l16) * KTOT + (k0 + kb);
        v16bf bf = cat8(*(const v8bf*)bp, *(const v8bf*)(bp + 16));
        #pragma unroll
        for (int ri = 0; ri < 2; ++ri)
          acc[ri][cj] = __builtin_amdgcn_wmma_f32_16x16x32_bf16(
              false, af[ri], false, bf, (short)0, acc[ri][cj], false, false);
      }
    }
    __syncthreads();                     // all x_t / old-state reads complete

    // start DMA of x_{t+1}; overlaps state update + GEMM2 + output stores
#if USE_TDM
    if (wave == 0 && t + 1 < TLEN) tdm_load_xt(Xbf, t + 1, Xs);
#else
    if (t + 1 < TLEN) {
      const int row = tid >> 4, k0c = (tid & 15) * 16;
      const __bf16* src = Xbf + (size_t)row * (TLEN * INDIM) + (size_t)(t + 1) * INDIM + k0c;
      *(v8bf*)&Xs[row * XSTR + k0c]     = *(const v8bf*)src;
      *(v8bf*)&Xs[row * XSTR + k0c + 8] = *(const v8bf*)(src + 8);
    }
#endif

    // ---- state update: S = tanh(P + b); capture input state of the final step ----
    #pragma unroll
    for (int ri = 0; ri < 2; ++ri) {
      #pragma unroll
      for (int cj = 0; cj < 2; ++cj) {
        #pragma unroll
        for (int r = 0; r < 8; ++r) {
          const int m   = rp * 32 + ri * 16 + r + kb;  // D-layout: M = r + 8*(lane>=16)
          const int col = c0 + cj * 16 + l16;
          const float s = tanhf(acc[ri][cj][r] + biasb[cj]);
          Ss[m * LDSS + col] = (__bf16)s;
          if (t == TLEN - 2)             // == prev_states[-1] of the scan
            stateout[(size_t)m * NST + col] = s;
        }
      }
    }
    __syncthreads();                     // new state visible

    // ---- GEMM2: Y = S @ [C | D]  (64x512 @ 512x512) ----
    v8f oac[2][2] = {};
    #pragma unroll
    for (int kc = 0; kc < NST / 32; ++kc) {
      const int k0 = kc * 32;
      v16bf af[2];
      #pragma unroll
      for (int ri = 0; ri < 2; ++ri) {
        const __bf16* sp = &Ss[(rp * 32 + ri * 16 + l16) * LDSS + k0 + kb];
        af[ri] = cat8(*(const v8bf*)sp, *(const v8bf*)(sp + 16));
      }
      #pragma unroll
      for (int cj = 0; cj < 2; ++cj) {
        const __bf16* bp = CDt + (size_t)(c0 + cj * 16 + l16) * NST + (k0 + kb);
        v16bf bf = cat8(*(const v8bf*)bp, *(const v8bf*)(bp + 16));
        #pragma unroll
        for (int ri = 0; ri < 2; ++ri)
          oac[ri][cj] = __builtin_amdgcn_wmma_f32_16x16x32_bf16(
              false, af[ri], false, bf, (short)0, oac[ri][cj], false, false);
      }
    }

    // ---- emit loc (cp 0-7) / softplus scale (cp 8-15), layout [2,B,T,OUT] ----
    if (c0 < ODIM) {
      #pragma unroll
      for (int ri = 0; ri < 2; ++ri)
        #pragma unroll
        for (int cj = 0; cj < 2; ++cj)
          #pragma unroll
          for (int r = 0; r < 8; ++r) {
            const int m   = rp * 32 + ri * 16 + r + kb;
            const int col = c0 + cj * 16 + l16;
            out[((size_t)m * TLEN + t) * ODIM + col] = oac[ri][cj][r] + biascd[cj];
          }
    } else {
      #pragma unroll
      for (int ri = 0; ri < 2; ++ri)
        #pragma unroll
        for (int cj = 0; cj < 2; ++cj)
          #pragma unroll
          for (int r = 0; r < 8; ++r) {
            const int m   = rp * 32 + ri * 16 + r + kb;
            const int col = c0 + cj * 16 + l16;
            const float v  = oac[ri][cj][r] + biascd[cj];
            const float sp = fmaxf(v, 0.0f) + log1pf(__expf(-fabsf(v)));  // stable softplus
            out[((size_t)(BSZ + m) * TLEN + t) * ODIM + (col - ODIM)] = sp;
          }
    }
  }
}

extern "C" void kernel_launch(void* const* d_in, const int* in_sizes, int n_in,
                              void* d_out, int out_size, void* d_ws, size_t ws_size,
                              hipStream_t stream) {
  const float* x  = (const float*)d_in[0];
  const float* A  = (const float*)d_in[1];
  const float* Bm = (const float*)d_in[2];
  const float* b  = (const float*)d_in[3];
  const float* C  = (const float*)d_in[4];
  const float* c  = (const float*)d_in[5];
  const float* D  = (const float*)d_in[6];
  const float* d  = (const float*)d_in[7];
  float* out = (float*)d_out;

  __bf16* Xbf = (__bf16*)d_ws;                       // 64x1024x256 bf16 = 32 MB
  __bf16* Wt  = Xbf + (size_t)BSZ * TLEN * INDIM;    // 512x768 bf16 = 768 KB
  __bf16* CDt = Wt + (size_t)NST * KTOT;             // 512x512 bf16 = 512 KB

  const size_t nx = (size_t)BSZ * TLEN * INDIM;
  pack_x_kernel <<<(unsigned)((nx + 255) / 256), 256, 0, stream>>>(x, Xbf);
  pack_w_kernel <<<(NST * KTOT + 255) / 256, 256, 0, stream>>>(A, Bm, Wt);
  pack_cd_kernel<<<(NST * NST  + 255) / 256, 256, 0, stream>>>(C, D, CDt);
  ssm_kernel    <<<1, 1024, 0, stream>>>(Xbf, b, c, d, Wt, CDt, out);
}
